// MyLSTMModel_4595615007139
// MI455X (gfx1250) — matrix operand order repaired and checked by reading, not compiled, not run on hardware
//
#include <hip/hip_runtime.h>

// LSTM: B=4096, T=512, IN=9, H=50, gates [i|f|g|o] each 50 wide.
// Fully fused, latency-optimized recurrence:
//  - all of this WG's x preloaded to LDS as f16 (no global ops in the time loop)
//  - double-buffered A staging -> ONE barrier per timestep
//  - x_t halves prefetched into REGISTERS one step ahead (ds_load latency hidden
//    under the WMMAs), so the pre-barrier section is just two ds_store_b16
//  - per-step GEMM: A = [h | x_t | 1] (K=64 padded), W = [w_hh^T; w_ih^T; bias],
//    8x v_wmma_f32_16x16x32_f16 per step (4 gates x 2 K-chunks), f32 accumulate
//  - B fragments loop-invariant in registers; c,h state in registers

#define B_TOT   4096
#define T_LEN   512
#define IN      9
#define H       50
#define NG      4
#define BM      16      // batch rows per workgroup (one WMMA M-tile)
#define THREADS 128     // 4 waves; wave w owns column subtile j = 16w..16w+15 of every gate

typedef __attribute__((ext_vector_type(16))) _Float16 v16h;
typedef __attribute__((ext_vector_type(8)))  _Float16 v8h;
typedef __attribute__((ext_vector_type(4)))  _Float16 v4h;
typedef __attribute__((ext_vector_type(8)))  float    v8f;

__device__ __forceinline__ float fast_sigmoid(float x) {
    // 1/(1+2^(-x*log2e))  -> v_exp_f32 + v_rcp_f32
    return __builtin_amdgcn_rcpf(1.0f + __builtin_amdgcn_exp2f(-1.44269504088896f * x));
}
__device__ __forceinline__ float fast_tanh(float x) {
    // 1 - 2/(1+2^(2x*log2e))
    return 1.0f - 2.0f * __builtin_amdgcn_rcpf(1.0f + __builtin_amdgcn_exp2f(2.88539008177793f * x));
}

__global__ __launch_bounds__(THREADS)
void lstm_fused_wmma(const float* __restrict__ x,
                     const float* __restrict__ w_ih,
                     const float* __restrict__ w_hh,
                     const float* __restrict__ b_ih,
                     const float* __restrict__ b_hh,
                     const float* __restrict__ fc_w,
                     const float* __restrict__ fc_b,
                     float* __restrict__ out)
{
    // 144 KB: this WG's entire x slice, f16, layout [m][t][k] (linear in global order)
    __shared__ __attribute__((aligned(16))) _Float16 xlds[BM * T_LEN * IN];
    // double-buffered WMMA A staging: [m][K], K=64 padded
    __shared__ __attribute__((aligned(16))) _Float16 Abuf[2][BM * 64];
    __shared__ float hfin[BM * H];

    const int tid  = threadIdx.x;
    const int lane = tid & 31;
    const int wave = tid >> 5;
    const int b0   = blockIdx.x * BM;

    // ---- preload x -> LDS (coalesced float4 global loads, f16 LDS stores) ----
    {
        const float4* xg = reinterpret_cast<const float4*>(x + (size_t)b0 * T_LEN * IN);
        const int nq = BM * T_LEN * IN / 4;           // 18432 float4
        for (int q = tid; q < nq; q += THREADS) {
            float4 v = xg[q];
            v4h h4;
            h4[0] = (_Float16)v.x; h4[1] = (_Float16)v.y;
            h4[2] = (_Float16)v.z; h4[3] = (_Float16)v.w;
            *reinterpret_cast<v4h*>(&xlds[q * 4]) = h4;
        }
    }

    // ---- init BOTH A buffers: zeros (h0=0, K-pad) + bias column K=59 = 1.0 ----
    for (int i = tid; i < 2 * BM * 64; i += THREADS)
        Abuf[0][i] = (_Float16)(((i & 63) == (H + IN)) ? 1.0f : 0.0f);

    // ---- build loop-invariant B fragments in registers ----
    // B tile (32x16 f16): lane L holds column n = L%16; halves q map to
    // K = 32*kc + 16*(L/16) + q (straight lane-half split).
    const int nloc = lane & 15;
    const int j    = wave * 16 + nloc;       // hidden-unit column this lane produces
    const int kgrp = (lane >> 4) * 16;

    v16h bfrag[NG][2];
    for (int g = 0; g < NG; ++g) {
        for (int kc = 0; kc < 2; ++kc) {
            v16h f;
            if (j < H) {
                const int row = g * H + j;   // row in the 4H=200 gate dimension
                for (int q = 0; q < 16; ++q) {
                    const int K = kc * 32 + kgrp + q;
                    float v;
                    if      (K < H)        v = w_hh[row * H + K];        // h part
                    else if (K < H + IN)   v = w_ih[row * IN + (K - H)]; // x part
                    else if (K == H + IN)  v = b_ih[row] + b_hh[row];    // bias via A==1
                    else                   v = 0.0f;                     // K pad
                    f[q] = (_Float16)v;
                }
            } else {
                for (int q = 0; q < 16; ++q) f[q] = (_Float16)0.0f;     // N pad (j>=50)
            }
            bfrag[g][kc] = f;
        }
    }

    // ---- per-lane recurrent state (C/D layout: lane -> n=j, VGPR r -> m = r + 8*(lane/16)) ----
    const int mrow = (lane >> 4) * 8;
    v8f c = {};
    v8f h = {};

    // A-fragment addressing (16-bit A 16x32 layout): lane L: m = L%16,
    // halves q -> K = 16*(L/16)+q (contiguous, matches B): two 16B runs.
    const int am  = lane & 15;
    const int ahb = (lane >> 4) * 16;

    // x staging: 144 elements (16 rows x 9); e0 = tid (always valid, 128 < 144),
    // e1 = tid + 128 (valid for tid < 16)
    const int e0 = tid, e1 = tid + THREADS;
    const int m0 = e0 / IN, k0 = e0 % IN;
    const int m1 = e1 / IN, k1 = e1 % IN;
    const bool p1 = (e1 < BM * IN);

    __syncthreads();   // x preload + Abuf init visible

    // register prefetch of x_t halves (t = 0)
    _Float16 xh0 = xlds[(m0 * T_LEN + 0) * IN + k0];
    _Float16 xh1 = p1 ? xlds[(m1 * T_LEN + 0) * IN + k1] : (_Float16)0.0f;

    auto step = [&](int t, _Float16* __restrict__ cur, _Float16* __restrict__ nxt) {
        // stage x_t from registers (no dependent LDS round-trip before the barrier)
        cur[m0 * 64 + H + k0] = xh0;
        if (p1) cur[m1 * 64 + H + k1] = xh1;
        __syncthreads();  // h_{t-1} (written end of prev step) + x_t visible

        // load A fragments: 4x ds_load_b128
        v8h lo0 = *reinterpret_cast<const v8h*>(&cur[am * 64 + ahb]);
        v8h hi0 = *reinterpret_cast<const v8h*>(&cur[am * 64 + 8 + ahb]);
        v8h lo1 = *reinterpret_cast<const v8h*>(&cur[am * 64 + 32 + ahb]);
        v8h hi1 = *reinterpret_cast<const v8h*>(&cur[am * 64 + 40 + ahb]);
        v16h a0, a1;
        #pragma unroll
        for (int q = 0; q < 8; ++q) {
            a0[q] = lo0[q]; a0[q + 8] = hi0[q];
            a1[q] = lo1[q]; a1[q + 8] = hi1[q];
        }

        // prefetch x_{t+1} into registers; latency hides under the WMMAs
        {
            const int tn = (t + 1 < T_LEN) ? (t + 1) : (T_LEN - 1);
            xh0 = xlds[(m0 * T_LEN + tn) * IN + k0];
            if (p1) xh1 = xlds[(m1 * T_LEN + tn) * IN + k1];
        }

        // gates = A[16x64] @ W[64x16] per gate: 2 chained K-chunks, 8 WMMAs
        v8f zero = {};
        v8f gi = __builtin_amdgcn_wmma_f32_16x16x32_f16(false, a0, false, bfrag[0][0], (short)0, zero, false, false);
        gi     = __builtin_amdgcn_wmma_f32_16x16x32_f16(false, a1, false, bfrag[0][1], (short)0, gi,   false, false);
        v8f gf = __builtin_amdgcn_wmma_f32_16x16x32_f16(false, a0, false, bfrag[1][0], (short)0, zero, false, false);
        gf     = __builtin_amdgcn_wmma_f32_16x16x32_f16(false, a1, false, bfrag[1][1], (short)0, gf,   false, false);
        v8f gg = __builtin_amdgcn_wmma_f32_16x16x32_f16(false, a0, false, bfrag[2][0], (short)0, zero, false, false);
        gg     = __builtin_amdgcn_wmma_f32_16x16x32_f16(false, a1, false, bfrag[2][1], (short)0, gg,   false, false);
        v8f go = __builtin_amdgcn_wmma_f32_16x16x32_f16(false, a0, false, bfrag[3][0], (short)0, zero, false, false);
        go     = __builtin_amdgcn_wmma_f32_16x16x32_f16(false, a1, false, bfrag[3][1], (short)0, go,   false, false);

        // elementwise LSTM cell update: i,f,g,o for (m,j) share lane/reg index
        #pragma unroll
        for (int r = 0; r < 8; ++r) {
            const float iv = fast_sigmoid(gi[r]);
            const float fv = fast_sigmoid(gf[r]);
            const float gv = fast_tanh(gg[r]);
            const float ov = fast_sigmoid(go[r]);
            const float cv = fv * c[r] + iv * gv;
            c[r] = cv;
            h[r] = ov * fast_tanh(cv);
        }

        // write h_t (f16) into the OTHER buffer (read next step) -- no 2nd barrier:
        // readers of `nxt` finished before this step's barrier.
        if (j < H) {
            #pragma unroll
            for (int r = 0; r < 8; ++r)
                nxt[(mrow + r) * 64 + j] = (_Float16)h[r];
        }
    };

    for (int t = 0; t < T_LEN; t += 2) {
        step(t,     &Abuf[0][0], &Abuf[1][0]);
        step(t + 1, &Abuf[1][0], &Abuf[0][0]);
    }

    // ---- final FC: out[b] = h_last[b,:] . fc_w + fc_b ----
    if (j < H) {
        #pragma unroll
        for (int r = 0; r < 8; ++r)
            hfin[(mrow + r) * H + j] = h[r];   // f32, no extra rounding
    }
    __syncthreads();
    if (tid < BM) {
        float s = fc_b[0];
        for (int q = 0; q < H; ++q) s += hfin[tid * H + q] * fc_w[q];
        out[b0 + tid] = s;
    }
}

extern "C" void kernel_launch(void* const* d_in, const int* in_sizes, int n_in,
                              void* d_out, int out_size, void* d_ws, size_t ws_size,
                              hipStream_t stream) {
    const float* x    = (const float*)d_in[0];
    const float* w_ih = (const float*)d_in[1];
    const float* w_hh = (const float*)d_in[2];
    const float* b_ih = (const float*)d_in[3];
    const float* b_hh = (const float*)d_in[4];
    const float* fc_w = (const float*)d_in[5];
    const float* fc_b = (const float*)d_in[6];
    float* out = (float*)d_out;

    dim3 grid(B_TOT / BM);   // 256 workgroups, one 16-row batch tile each
    dim3 block(THREADS);     // 4 waves
    lstm_fused_wmma<<<grid, block, 0, stream>>>(x, w_ih, w_hh, b_ih, b_hh, fc_w, fc_b, out);
}
